// GNNStack_69260642615296
// MI455X (gfx1250) — matrix operand order repaired and checked by reading, compiled, not run on hardware
//
#include <hip/hip_runtime.h>
#include <hip/hip_bf16.h>

typedef __attribute__((ext_vector_type(16))) __bf16 bf16x16;
typedef __attribute__((ext_vector_type(8)))  float  f32x8;

#define N_NODES 50000
#define N_EDGES 800000
#define D_INF   128
#define HEADS   8
#define CH      32
#define D_HIDF  256
#define D_OUTF  40
#define D_OUTP  48

// ---------- helpers ----------
__device__ __forceinline__ unsigned encf(float f) {
    unsigned u = __float_as_uint(f);
    return ((int)u < 0) ? ~u : (u | 0x80000000u);
}
__device__ __forceinline__ float decf(unsigned u) {
    return __uint_as_float(((int)u < 0) ? (u & 0x7FFFFFFFu) : ~u);
}
#define ENC_NEG_INF 0x007FFFFFu   // encf(-inf)

// ---------- conversions ----------
__global__ void k_f32_to_bf16(const float* __restrict__ in, __bf16* __restrict__ out, int n) {
    int i = blockIdx.x * blockDim.x + threadIdx.x;
    if (i < n) out[i] = (__bf16)in[i];
}

// rows >= rowsReal are zero-filled (padding for Wp2)
__global__ void k_f32_to_bf16_padrows(const float* __restrict__ in, __bf16* __restrict__ out,
                                      int rowsReal, int rowsPad, int K) {
    int i = blockIdx.x * blockDim.x + threadIdx.x;
    if (i >= rowsPad * K) return;
    int r = i / K;
    out[i] = (r < rowsReal) ? (__bf16)in[i] : (__bf16)0.0f;
}

// ---------- WMMA GEMM: D[M,Nreal] = A[M,K](bf16) * W[Npad,K]^T(bf16) + bias ----------
__global__ __launch_bounds__(256)
void k_wmma_gemm(const __bf16* __restrict__ A, const __bf16* __restrict__ W,
                 const float* __restrict__ bias,
                 float* __restrict__ D, __bf16* __restrict__ Dbf,
                 int M, int K, int Npad, int Nreal, int relu)
{
    const int lane = threadIdx.x & 31;
    const int wave = threadIdx.x >> 5;
    const int tilesN = Npad >> 4;
    const int tile = blockIdx.x * 8 + wave;
    const int totalTiles = (M >> 4) * tilesN;
    if (tile >= totalTiles) return;

    const int tm = tile / tilesN;
    const int tn = tile - tm * tilesN;
    const int half = lane >> 4;     // 0: low K-half lanes, 1: high
    const int r    = lane & 15;

    const int arow = tm * 16 + r;   // A row held by this lane
    const int bcol = tn * 16 + r;   // B column held by this lane

    f32x8 acc;
    {
        float bv = (bcol < Nreal) ? bias[bcol] : 0.0f;
        #pragma unroll
        for (int j = 0; j < 8; ++j) acc[j] = bv;
    }

    const __bf16* aptr = A + (size_t)arow * K;
    const __bf16* bptr = W + (size_t)bcol * K;

    #pragma unroll 4
    for (int k0 = 0; k0 < K; k0 += 32) {
        union { bf16x16 v; uint4 u[2]; } a, b;
        // A 16x32 bf16 layout: lanes0-15 K={0..7,16..23}, lanes16-31 K={8..15,24..31}
        a.u[0] = *(const uint4*)(aptr + k0 + half * 8);
        a.u[1] = *(const uint4*)(aptr + k0 + 16 + half * 8);
        // B 32x16 bf16 layout: lanes0-15 K={0..7,16..23} col r, lanes16-31 K={8..15,24..31} col r
        b.u[0] = *(const uint4*)(bptr + k0 + half * 8);
        b.u[1] = *(const uint4*)(bptr + k0 + 16 + half * 8);
        acc = __builtin_amdgcn_wmma_f32_16x16x32_bf16(false, a.v, false, b.v,
                                                      (short)0, acc, false, false);
    }

    // C/D layout: VGPR j -> row (j + half*8), col r
    #pragma unroll
    for (int j = 0; j < 8; ++j) {
        if (bcol < Nreal) {
            int row = tm * 16 + half * 8 + j;
            float v = acc[j];
            if (relu) v = v > 0.0f ? v : 0.0f;
            if (D)   D  [(size_t)row * Nreal + bcol] = v;
            if (Dbf) Dbf[(size_t)row * Nreal + bcol] = (__bf16)v;
        }
    }
}

// ---------- per-node attention scalars ----------
__global__ void k_att_scalars(const float* __restrict__ h, const float* __restrict__ att,
                              float* __restrict__ adst, float* __restrict__ asrc)
{
    int idx = blockIdx.x * blockDim.x + threadIdx.x;     // over N*H
    if (idx >= N_NODES * HEADS) return;
    int n = idx >> 3, hh = idx & 7;
    const float* hp = h   + (size_t)n * D_HIDF + hh * CH;
    const float* ad = att + hh * (2 * CH);
    const float* as = ad + CH;
    float sd = 0.0f, ss = 0.0f;
    #pragma unroll
    for (int c = 0; c < CH; ++c) { float v = hp[c]; sd += v * ad[c]; ss += v * as[c]; }
    adst[idx] = sd;
    asrc[idx] = ss;
}

// ---------- init emax/denom/agg ----------
__global__ void k_init_layer(unsigned* __restrict__ emax, float* __restrict__ denom,
                             float* __restrict__ agg)
{
    int i = blockIdx.x * blockDim.x + threadIdx.x;       // over N*256
    if (i >= N_NODES * D_HIDF) return;
    agg[i] = 0.0f;
    if (i < N_NODES * HEADS) { emax[i] = ENC_NEG_INF; denom[i] = 0.0f; }
}

// ---------- edge pass 1: logits + segment max ----------
__global__ void k_edge_logits(const int* __restrict__ ei, const float* __restrict__ adst,
                              const float* __restrict__ asrc, float* __restrict__ ew,
                              unsigned* __restrict__ emax)
{
    int e = blockIdx.x * blockDim.x + threadIdx.x;
    if (e >= N_EDGES) return;
    int s = ei[e], d = ei[N_EDGES + e];
    const float4* Ad = (const float4*)(adst + (size_t)d * 8);
    const float4* As = (const float4*)(asrc + (size_t)s * 8);
    float4 d0 = Ad[0], d1 = Ad[1], s0 = As[0], s1 = As[1];
    float vals[8] = { d0.x + s0.x, d0.y + s0.y, d0.z + s0.z, d0.w + s0.w,
                      d1.x + s1.x, d1.y + s1.y, d1.z + s1.z, d1.w + s1.w };
    #pragma unroll
    for (int hh = 0; hh < 8; ++hh) {
        float v = vals[hh];
        v = v > 0.0f ? v : 0.2f * v;                 // leaky_relu(0.2)
        ew[(size_t)e * 8 + hh] = v;
        atomicMax(&emax[(size_t)d * 8 + hh], encf(v));
    }
}

// ---------- edge pass 2: exp + segment sum ----------
__global__ void k_edge_weights(const int* __restrict__ ei, float* __restrict__ ew,
                               const unsigned* __restrict__ emax, float* __restrict__ denom)
{
    int e = blockIdx.x * blockDim.x + threadIdx.x;
    if (e >= N_EDGES) return;
    int d = ei[N_EDGES + e];
    #pragma unroll
    for (int hh = 0; hh < 8; ++hh) {
        float m = decf(emax[(size_t)d * 8 + hh]);
        float w = __expf(ew[(size_t)e * 8 + hh] - m);
        ew[(size_t)e * 8 + hh] = w;
        atomicAdd(&denom[(size_t)d * 8 + hh], w);
    }
}

// ---------- edge pass 3: weighted aggregation (warp per edge) ----------
__global__ __launch_bounds__(256)
void k_edge_aggregate(const int* __restrict__ ei, const float* __restrict__ h,
                      const float* __restrict__ ew, const float* __restrict__ denom,
                      float* __restrict__ agg)
{
    int gw   = (blockIdx.x * blockDim.x + threadIdx.x) >> 5;  // edge id
    int lane = threadIdx.x & 31;
    if (gw >= N_EDGES) return;
    int s = __builtin_amdgcn_readfirstlane(ei[gw]);
    int d = __builtin_amdgcn_readfirstlane(ei[N_EDGES + gw]);
    int hh = lane >> 2;                                       // 8 channels -> head = lane/4
    float alpha = ew[(size_t)gw * 8 + hh] / (denom[(size_t)d * 8 + hh] + 1e-16f);
    const float4* hp = (const float4*)(h + (size_t)s * D_HIDF + lane * 8);
    float4 v0 = hp[0], v1 = hp[1];
    float* op = agg + (size_t)d * D_HIDF + lane * 8;
    atomicAdd(op + 0, v0.x * alpha);
    atomicAdd(op + 1, v0.y * alpha);
    atomicAdd(op + 2, v0.z * alpha);
    atomicAdd(op + 3, v0.w * alpha);
    atomicAdd(op + 4, v1.x * alpha);
    atomicAdd(op + 5, v1.y * alpha);
    atomicAdd(op + 6, v1.z * alpha);
    atomicAdd(op + 7, v1.w * alpha);
}

// ---------- epilogue: bias + relu + bf16 for next GEMM ----------
__global__ void k_bias_relu_bf16(const float* __restrict__ agg, const float* __restrict__ bias,
                                 __bf16* __restrict__ xb)
{
    int i = blockIdx.x * blockDim.x + threadIdx.x;       // over N*256
    if (i >= N_NODES * D_HIDF) return;
    float v = agg[i] + bias[i & (D_HIDF - 1)];
    xb[i] = (__bf16)(v > 0.0f ? v : 0.0f);
}

// ---------- log_softmax over 40 classes, in-place ----------
__global__ void k_log_softmax(float* __restrict__ out)
{
    int n = blockIdx.x * blockDim.x + threadIdx.x;
    if (n >= N_NODES) return;
    float* p = out + (size_t)n * D_OUTF;
    float v[D_OUTF];
    float m = -INFINITY;
    #pragma unroll
    for (int i = 0; i < D_OUTF; ++i) { v[i] = p[i]; m = fmaxf(m, v[i]); }
    float s = 0.0f;
    #pragma unroll
    for (int i = 0; i < D_OUTF; ++i) s += __expf(v[i] - m);
    float l = m + __logf(s);
    #pragma unroll
    for (int i = 0; i < D_OUTF; ++i) p[i] = v[i] - l;
}

// =====================================================================
extern "C" void kernel_launch(void* const* d_in, const int* in_sizes, int n_in,
                              void* d_out, int out_size, void* d_ws, size_t ws_size,
                              hipStream_t stream)
{
    const float* x    = (const float*)d_in[0];
    const int*   ei   = (const int*)  d_in[1];
    const float* W1   = (const float*)d_in[2];
    const float* b1   = (const float*)d_in[3];
    const float* att1 = (const float*)d_in[4];
    const float* bi1  = (const float*)d_in[5];
    const float* W2   = (const float*)d_in[6];
    const float* b2   = (const float*)d_in[7];
    const float* att2 = (const float*)d_in[8];
    const float* bi2  = (const float*)d_in[9];
    const float* Wp1  = (const float*)d_in[10];
    const float* bp1  = (const float*)d_in[11];
    const float* Wp2  = (const float*)d_in[12];
    const float* bp2  = (const float*)d_in[13];
    float* out = (float*)d_out;

    // ---- workspace layout ----
    char* ws = (char*)d_ws;
    size_t off = 0;
    auto alloc = [&](size_t bytes) -> void* {
        void* p = ws + off;
        off = (off + bytes + 255) & ~(size_t)255;
        return p;
    };
    __bf16* xb    = (__bf16*)alloc((size_t)N_NODES * D_INF  * 2);
    __bf16* actb  = (__bf16*)alloc((size_t)N_NODES * D_HIDF * 2);  // x1b then x2b
    __bf16* p1b   = (__bf16*)alloc((size_t)N_NODES * D_HIDF * 2);
    float*  h     = (float*) alloc((size_t)N_NODES * D_HIDF * 4);
    float*  agg   = (float*) alloc((size_t)N_NODES * D_HIDF * 4);
    float*  adst  = (float*) alloc((size_t)N_NODES * HEADS  * 4);
    float*  asrc  = (float*) alloc((size_t)N_NODES * HEADS  * 4);
    unsigned* emax= (unsigned*)alloc((size_t)N_NODES * HEADS * 4);
    float*  denom = (float*) alloc((size_t)N_NODES * HEADS  * 4);
    float*  ew    = (float*) alloc((size_t)N_EDGES * HEADS  * 4);
    __bf16* W1b   = (__bf16*)alloc((size_t)D_HIDF * D_INF  * 2);
    __bf16* W2b   = (__bf16*)alloc((size_t)D_HIDF * D_HIDF * 2);
    __bf16* Wp1b  = (__bf16*)alloc((size_t)D_HIDF * D_HIDF * 2);
    __bf16* Wp2b  = (__bf16*)alloc((size_t)D_OUTP * D_HIDF * 2);
    (void)ws_size; (void)n_in; (void)in_sizes; (void)out_size;

    const int T = 256;
    auto blocks = [](long n, int t) { return (int)((n + t - 1) / t); };

    // ---- weight/input conversions ----
    k_f32_to_bf16<<<blocks((long)N_NODES * D_INF, T), T, 0, stream>>>(x, xb, N_NODES * D_INF);
    k_f32_to_bf16<<<blocks(D_HIDF * D_INF, T), T, 0, stream>>>(W1, W1b, D_HIDF * D_INF);
    k_f32_to_bf16<<<blocks(D_HIDF * D_HIDF, T), T, 0, stream>>>(W2, W2b, D_HIDF * D_HIDF);
    k_f32_to_bf16<<<blocks(D_HIDF * D_HIDF, T), T, 0, stream>>>(Wp1, Wp1b, D_HIDF * D_HIDF);
    k_f32_to_bf16_padrows<<<blocks(D_OUTP * D_HIDF, T), T, 0, stream>>>(Wp2, Wp2b, D_OUTF, D_OUTP, D_HIDF);

    const int gemmBlocksHid = blocks((N_NODES / 16) * (D_HIDF / 16), 8);
    const int gemmBlocksOut = blocks((N_NODES / 16) * (D_OUTP / 16), 8);
    const int nh  = blocks((long)N_NODES * D_HIDF, T);
    const int nhe = blocks(N_NODES * HEADS, T);
    const int ne  = blocks(N_EDGES, T);
    const int new32 = blocks((long)N_EDGES * 32, T);

    // ================= GAT layer 1 =================
    k_wmma_gemm<<<gemmBlocksHid, T, 0, stream>>>(xb, W1b, b1, h, (/*Dbf*/__bf16*)nullptr,
                                                 N_NODES, D_INF, D_HIDF, D_HIDF, 0);
    k_att_scalars<<<nhe, T, 0, stream>>>(h, att1, adst, asrc);
    k_init_layer<<<nh, T, 0, stream>>>(emax, denom, agg);
    k_edge_logits<<<ne, T, 0, stream>>>(ei, adst, asrc, ew, emax);
    k_edge_weights<<<ne, T, 0, stream>>>(ei, ew, emax, denom);
    k_edge_aggregate<<<new32, T, 0, stream>>>(ei, h, ew, denom, agg);
    k_bias_relu_bf16<<<nh, T, 0, stream>>>(agg, bi1, actb);

    // ================= GAT layer 2 =================
    k_wmma_gemm<<<gemmBlocksHid, T, 0, stream>>>(actb, W2b, b2, h, (__bf16*)nullptr,
                                                 N_NODES, D_HIDF, D_HIDF, D_HIDF, 0);
    k_att_scalars<<<nhe, T, 0, stream>>>(h, att2, adst, asrc);
    k_init_layer<<<nh, T, 0, stream>>>(emax, denom, agg);
    k_edge_logits<<<ne, T, 0, stream>>>(ei, adst, asrc, ew, emax);
    k_edge_weights<<<ne, T, 0, stream>>>(ei, ew, emax, denom);
    k_edge_aggregate<<<new32, T, 0, stream>>>(ei, h, ew, denom, agg);
    k_bias_relu_bf16<<<nh, T, 0, stream>>>(agg, bi2, actb);   // actb = x2 (bf16)

    // ================= post-MP MLP =================
    k_wmma_gemm<<<gemmBlocksHid, T, 0, stream>>>(actb, Wp1b, bp1, h, p1b,
                                                 N_NODES, D_HIDF, D_HIDF, D_HIDF, 0);
    k_wmma_gemm<<<gemmBlocksOut, T, 0, stream>>>(p1b, Wp2b, bp2, out, (__bf16*)nullptr,
                                                 N_NODES, D_HIDF, D_OUTP, D_OUTF, 0);
    k_log_softmax<<<blocks(N_NODES, T), T, 0, stream>>>(out);
}
